// H_H_EdgeApplyMoudle_47682726921127
// MI455X (gfx1250) — matrix operand order repaired and checked by reading, not compiled, not run on hardware
//
#include <hip/hip_runtime.h>

// ---- vector types for WMMA fragments -------------------------------------
typedef __attribute__((ext_vector_type(16))) __bf16 v16bf;
typedef __attribute__((ext_vector_type(8)))  __bf16 v8bf;
typedef __attribute__((ext_vector_type(4)))  __bf16 v4bf;
typedef __attribute__((ext_vector_type(8)))  float  v8f;

#define D_NODE 1024
#define D_HID  1024
#define K1     2048      // gemm1 K (= 2*D_NODE), also row stride of sA
#define E_TILE 32        // edges per workgroup
#define CH_K   256       // k-columns per async gather chunk
#define NCHUNK (K1 / CH_K)   // 8 chunks; chunks 0-3 = src half, 4-7 = dst half

// ---------------------------------------------------------------------------
// A-fragment loader: 16x32 bf16 tile, contiguous k order (B uses the same):
//   lanes 0-15 : row M=lane,    K = {0..15}
//   lanes 16-31: row M=lane-16, K = {16..31}
// rowPtr points at &tile[mBase + (lane&15)][kBase]
__device__ __forceinline__ v16bf load_a_frag(const __bf16* rowPtr, int laneHi) {
  const __bf16* p = rowPtr + (laneHi ? 16 : 0);
  v8bf lo = *(const v8bf*)(p);        // K  base.. base+7
  v8bf hi = *(const v8bf*)(p + 8);    // K 8+base..8+base+7
  return __builtin_shufflevector(lo, hi, 0,1,2,3,4,5,6,7,8,9,10,11,12,13,14,15);
}

__device__ __forceinline__ v8f wmma_bf16(v16bf a, v16bf b, v8f c) {
  // D = A(16x32) * B(32x16) + C, f32 accumulate
  return __builtin_amdgcn_wmma_f32_16x16x32_bf16(
      /*neg_a=*/false, a, /*neg_b=*/false, b,
      /*c_mod=*/(short)0, c, /*reuse_a=*/false, /*reuse_b=*/false);
}

// ---------------------------------------------------------------------------
// Pass 0: fp32 -> bf16 with transpose, so B fragments are contiguous per lane.
//   W1T[n][k] = W1[k][n]  (1024 x 2048)
//   W2T[n][k] = W2[k][n]  (1024 x 1024)
__global__ __launch_bounds__(256) void convert_weights_kernel(
    const float* __restrict__ W1, const float* __restrict__ W2,
    __bf16* __restrict__ W1T, __bf16* __restrict__ W2T)
{
  const int i  = blockIdx.x * 256 + threadIdx.x;
  const int n1 = K1 * D_HID;       // 2048*1024
  const int n2 = D_HID * D_HID;    // 1024*1024
  if (i < n1) {
    const int n = i >> 11;         // / 2048
    const int k = i & 2047;
    W1T[i] = (__bf16)W1[(size_t)k * D_HID + n];
  } else if (i < n1 + n2) {
    const int j = i - n1;
    const int n = j >> 10;
    const int k = j & 1023;
    W2T[(size_t)j] = (__bf16)W2[(size_t)k * D_HID + n];
    (void)n; (void)k;
  }
}

// ---------------------------------------------------------------------------
// Fused edge-MLP: async gather -> GEMM1(+bias+ReLU, h in LDS) -> GEMM2.
// Block: 256 threads (8 wave32), 32 edges per block.
// Dynamic LDS layout (262,656 B total):
//   [0,512)            : 64 x i64 gathered node indices
//   [512, +131072)     : sA  32 x 2048 bf16  (concat features)
//   [.., +65536)       : sH  32 x 1024 bf16  (hidden activations)
//   [.., +65536)       : sF  2 x (32 x 256) fp32 async-gather staging buffers
__global__ __launch_bounds__(256) void edge_mlp_fused_kernel(
    const float*     __restrict__ nf,    // [50000,1024] fp32
    const long long* __restrict__ src,   // [65536] i64
    const long long* __restrict__ dst,   // [65536] i64
    const float*     __restrict__ b1,    // [1024]
    const float*     __restrict__ b2,    // [1024]
    const __bf16*    __restrict__ W1T,   // [1024][2048] bf16
    const __bf16*    __restrict__ W2T,   // [1024][1024] bf16
    float*           __restrict__ out)   // [65536,1024] fp32
{
  extern __shared__ char smem[];
  long long* sIdx = (long long*)smem;                              // 64 entries
  __bf16*    sA   = (__bf16*)(smem + 512);                         // 32 x 2048
  __bf16*    sH   = (__bf16*)(smem + 512 + E_TILE * K1 * 2);       // 32 x 1024
  float*     sF   = (float*)(smem + 512 + E_TILE * K1 * 2
                                       + E_TILE * D_HID * 2);      // 2 x 8192 fp32

  const int tid = threadIdx.x;
  const int e0  = blockIdx.x * E_TILE;

  if (tid < E_TILE)            sIdx[tid] = src[e0 + tid];
  else if (tid < 2 * E_TILE)   sIdx[tid] = dst[e0 + tid - E_TILE];
  __syncthreads();

  // ---- Stage 1: double-buffered async gather (HBM -> LDS, ASYNCcnt) ------
  // Chunk = 32 rows x 256 k fp32 (32 KB) = 2048 float4 / 256 threads = 8 each.
  // Each thread later converts exactly the float4s it issued, so per-wave
  // s_wait_asynccnt is the only ordering needed until the final barrier.
  auto issue_chunk = [&](int c, int buf) {
    const int half    = (c * CH_K) >= D_NODE;           // uniform per chunk
    const int colBase = c * CH_K - (half ? D_NODE : 0);
#pragma unroll
    for (int j = 0; j < 8; ++j) {
      const int f4 = j * 256 + tid;                     // 0..2047
      const int r  = f4 >> 6;                           // 64 float4 per row
      const int kc = (f4 & 63) << 2;
      const long long node = sIdx[(half ? E_TILE : 0) + r];
      const float* gp = nf + (size_t)node * D_NODE + colBase + kc;
      const unsigned ldsAddr =
          (unsigned)(unsigned long long)(sF + buf * (E_TILE * CH_K) + r * CH_K + kc);
      asm volatile("global_load_async_to_lds_b128 %0, %1, off"
                   :: "v"(ldsAddr), "v"(gp) : "memory");
    }
  };
  auto convert_chunk = [&](int c, int buf) {
#pragma unroll
    for (int j = 0; j < 8; ++j) {
      const int f4 = j * 256 + tid;
      const int r  = f4 >> 6;
      const int kc = (f4 & 63) << 2;
      const float4 v = *(const float4*)(sF + buf * (E_TILE * CH_K) + r * CH_K + kc);
      v4bf o = { (__bf16)v.x, (__bf16)v.y, (__bf16)v.z, (__bf16)v.w };
      *(v4bf*)(sA + r * K1 + c * CH_K + kc) = o;
    }
  };

  issue_chunk(0, 0);
  for (int c = 0; c < NCHUNK; ++c) {
    if (c + 1 < NCHUNK) {
      issue_chunk(c + 1, (c + 1) & 1);
      asm volatile("s_wait_asynccnt 0x8" ::: "memory");  // chunk c landed
    } else {
      asm volatile("s_wait_asynccnt 0x0" ::: "memory");
    }
    convert_chunk(c, c & 1);
  }
  __syncthreads();

  const int lane   = tid & 31;
  const int lane15 = lane & 15;
  const int laneHi = lane >> 4;
  const int w      = tid >> 5;      // wave id 0..7; each wave owns 8 N-tiles

  // ---- Stage 2: h = relu(A @ W1 + b1) -> sH (bf16), M=32 N=1024 K=2048 ----
  for (int nt = 0; nt < 8; ++nt) {
    const int nBase = (w * 8 + nt) * 16;
    // B fragment: column (nBase+lane15) of W1, contiguous in K;
    // lanes 0-15 take K kb..kb+15, lanes 16-31 take K kb+16..kb+31.
    const __bf16* bcol = W1T + (size_t)(nBase + lane15) * K1 + laneHi * 16;
    const __bf16* a0p  = sA + (size_t)(0  + lane15) * K1;
    const __bf16* a1p  = sA + (size_t)(16 + lane15) * K1;
    v8f acc0 = {}; v8f acc1 = {};
    v16bf bcur = *(const v16bf*)(bcol);                 // software pipeline B
#pragma unroll 4
    for (int k = 0; k < K1 - 32; k += 32) {
      const v16bf bnext = *(const v16bf*)(bcol + k + 32);
      const v16bf a0 = load_a_frag(a0p + k, laneHi);
      const v16bf a1 = load_a_frag(a1p + k, laneHi);
      acc0 = wmma_bf16(a0, bcur, acc0);
      acc1 = wmma_bf16(a1, bcur, acc1);
      bcur = bnext;
    }
    {
      const v16bf a0 = load_a_frag(a0p + K1 - 32, laneHi);
      const v16bf a1 = load_a_frag(a1p + K1 - 32, laneHi);
      acc0 = wmma_bf16(a0, bcur, acc0);
      acc1 = wmma_bf16(a1, bcur, acc1);
    }
    const float bb = b1[nBase + lane15];
#pragma unroll
    for (int v = 0; v < 8; ++v) {
      const int m = v + 8 * laneHi;          // C layout: VGPR v -> row v / v+8
      sH[(size_t)(m)      * D_HID + nBase + lane15] = (__bf16)fmaxf(acc0[v] + bb, 0.0f);
      sH[(size_t)(16 + m) * D_HID + nBase + lane15] = (__bf16)fmaxf(acc1[v] + bb, 0.0f);
    }
  }
  __syncthreads();

  // ---- Stage 3: out = h @ W2 + b2, M=32 N=1024 K=1024 ---------------------
  for (int nt = 0; nt < 8; ++nt) {
    const int nBase = (w * 8 + nt) * 16;
    const __bf16* bcol = W2T + (size_t)(nBase + lane15) * D_HID + laneHi * 16;
    const __bf16* a0p  = sH + (size_t)(0  + lane15) * D_HID;
    const __bf16* a1p  = sH + (size_t)(16 + lane15) * D_HID;
    v8f acc0 = {}; v8f acc1 = {};
    v16bf bcur = *(const v16bf*)(bcol);
#pragma unroll 4
    for (int k = 0; k < D_HID - 32; k += 32) {
      const v16bf bnext = *(const v16bf*)(bcol + k + 32);
      const v16bf a0 = load_a_frag(a0p + k, laneHi);
      const v16bf a1 = load_a_frag(a1p + k, laneHi);
      acc0 = wmma_bf16(a0, bcur, acc0);
      acc1 = wmma_bf16(a1, bcur, acc1);
      bcur = bnext;
    }
    {
      const v16bf a0 = load_a_frag(a0p + D_HID - 32, laneHi);
      const v16bf a1 = load_a_frag(a1p + D_HID - 32, laneHi);
      acc0 = wmma_bf16(a0, bcur, acc0);
      acc1 = wmma_bf16(a1, bcur, acc1);
    }
    const float bb = b2[nBase + lane15];
#pragma unroll
    for (int v = 0; v < 8; ++v) {
      const int m = v + 8 * laneHi;
      out[(size_t)(e0 + m)      * D_HID + nBase + lane15] = acc0[v] + bb;
      out[(size_t)(e0 + 16 + m) * D_HID + nBase + lane15] = acc1[v] + bb;
    }
  }
}

// ---------------------------------------------------------------------------
extern "C" void kernel_launch(void* const* d_in, const int* in_sizes, int n_in,
                              void* d_out, int out_size, void* d_ws, size_t ws_size,
                              hipStream_t stream) {
  const float*     nf  = (const float*)d_in[0];
  const long long* src = (const long long*)d_in[1];   // jnp.int64
  const long long* dst = (const long long*)d_in[2];   // jnp.int64
  const float*     W1  = (const float*)d_in[3];
  const float*     b1  = (const float*)d_in[4];
  const float*     W2  = (const float*)d_in[5];
  const float*     b2  = (const float*)d_in[6];
  float*           out = (float*)d_out;

  // workspace: W1T bf16 (4 MB) | W2T bf16 (2 MB)
  __bf16* W1T = (__bf16*)d_ws;
  __bf16* W2T = W1T + (size_t)K1 * D_HID;

  const int convTotal = K1 * D_HID + D_HID * D_HID;   // 3,145,728
  convert_weights_kernel<<<(convTotal + 255) / 256, 256, 0, stream>>>(W1, W2, W1T, W2T);

  const size_t smemBytes = 512 + (size_t)E_TILE * K1 * 2      // sA
                               + (size_t)E_TILE * D_HID * 2   // sH
                               + 2 * (size_t)E_TILE * CH_K * 4; // sF (double buf)
  const int    nBlocks   = 65536 / E_TILE;            // 2048
  edge_mlp_fused_kernel<<<nBlocks, 256, smemBytes, stream>>>(
      nf, src, dst, b1, b2, W1T, W2T, out);
}
